// Transformer_block_23922967838738
// MI455X (gfx1250) — matrix operand order repaired and checked by reading, not compiled, not run on hardware
//
#include <hip/hip_runtime.h>
#include <stdint.h>

#define BATCH  4
#define SEQ    2048
#define DMODEL 1024
#define NHEAD  16
#define DHEAD  64
#define FFDIM  256
#define MROWS  (BATCH * SEQ)   // 8192

typedef __bf16 bf16;
typedef __attribute__((ext_vector_type(16))) __bf16 v16bf;
typedef __attribute__((ext_vector_type(8)))  float  v8f;
typedef __attribute__((ext_vector_type(4)))  float  f32x4;
typedef __attribute__((ext_vector_type(4)))  unsigned int u32x4;
typedef __attribute__((ext_vector_type(4)))  int    v4i;

// ---------------- async global->LDS (CDNA5), guarded ----------------
#if defined(__AMDGCN__) && defined(__has_builtin)
#if __has_builtin(__builtin_amdgcn_global_load_async_to_lds_b128) && \
    __has_builtin(__builtin_amdgcn_s_wait_asynccnt)
#define USE_ASYNC 1
#endif
#endif
#ifndef USE_ASYNC
#define USE_ASYNC 0
#endif

#if USE_ASYNC
// codegen-confirmed: emits global_load_async_to_lds_b128 / s_wait_asynccnt
#define ASYNC_B128(g, s) \
  __builtin_amdgcn_global_load_async_to_lds_b128((v4i*)(g), (v4i*)(s), 0, 0)
#define ASYNC_WAIT(n)    __builtin_amdgcn_s_wait_asynccnt(n)
#else
#define ASYNC_B128(g, s) (*(u32x4*)(s) = *(const u32x4*)(g))
#define ASYNC_WAIT(n)    ((void)0)
#endif

// ---------------- WMMA helpers ----------------
__device__ static inline v8f v8f_zero() {
  v8f z;
#pragma unroll
  for (int i = 0; i < 8; ++i) z[i] = 0.0f;
  return z;
}

__device__ static inline v8f wmma_bf16(v16bf a, v16bf b, v8f c) {
  return __builtin_amdgcn_wmma_f32_16x16x32_bf16(false, a, false, b, (short)0, c,
                                                 false, false);
}

// A-fragment (16x32, M x K), ISA table layout:
// lane<16: row=r0+l, k in [0,8) and [16,24); lane>=16: k in [8,16) and [24,32).
__device__ static inline v16bf load_frag_a(const bf16* tile, int r0, int c0, int ld) {
  const int lane = (int)(threadIdx.x & 31);
  const int l = lane & 15, hf = lane >> 4;
  const bf16* p = tile + (size_t)(r0 + l) * ld + c0 + hf * 8;
  union { v16bf v; u32x4 q[2]; } u;
  u.q[0] = *(const u32x4*)p;
  u.q[1] = *(const u32x4*)(p + 16);
  return u.v;
}

// B-fragment (32x16, K x N) staged as B^T rows ([N][K] tile):
// lane<16: col n=r0+l, k in [0,8) and [16,24); lane>=16: k in [8,16) and [24,32).
__device__ static inline v16bf load_frag_b(const bf16* tile, int r0, int c0, int ld) {
  const int lane = (int)(threadIdx.x & 31);
  const int l = lane & 15, hf = lane >> 4;
  const bf16* p = tile + (size_t)(r0 + l) * ld + c0 + hf * 8;
  union { v16bf v; u32x4 q[2]; } u;
  u.q[0] = *(const u32x4*)p;
  u.q[1] = *(const u32x4*)(p + 16);
  return u.v;
}

// ---------------- conversion kernels ----------------
__global__ void cvt_bf16_kernel(const float* __restrict__ src, bf16* __restrict__ dst,
                                int n) {
  int i = (blockIdx.x * blockDim.x + threadIdx.x) * 4;
  if (i + 3 < n) {
    f32x4 v = *(const f32x4*)(src + i);
    union { bf16 e[4]; unsigned long long u; } o;
    o.e[0] = (bf16)v[0]; o.e[1] = (bf16)v[1];
    o.e[2] = (bf16)v[2]; o.e[3] = (bf16)v[3];
    *(unsigned long long*)(dst + i) = o.u;
  }
}

// Wt[n*K + k] = (bf16) W[k*N + n]   (W is [K,N] row-major fp32)
__global__ void cvt_t_kernel(const float* __restrict__ W, bf16* __restrict__ Wt,
                             int K, int N) {
  int id = blockIdx.x * blockDim.x + threadIdx.x;
  int kk = id / N, n = id % N;
  Wt[(size_t)n * K + kk] = (bf16)W[(size_t)kk * N + n];
}

// ---------------- generic WMMA GEMM: C = A[M,K] x Wt[N,K]^T + bias -----------
// 128x128 block tile, BK=32, 256 threads (8 waves, 4x2), wave tile 32x64,
// double-buffered async-LDS staging. 8 WMMA vs 6 fragment loads per k-step.
__global__ __launch_bounds__(256) void gemm_bf16_wmma(
    const bf16* __restrict__ A, const bf16* __restrict__ Wt,
    const float* __restrict__ bias, float* __restrict__ outF,
    bf16* __restrict__ outB, int Mrows, int N, int K, int act) {
  constexpr int LD = 40;  // 32 + 8 pad (elems)
  __shared__ __attribute__((aligned(16))) bf16 sA[2][128 * LD];
  __shared__ __attribute__((aligned(16))) bf16 sB[2][128 * LD];

  const int t = (int)threadIdx.x;
  const int lane = t & 31;
  const int w = t >> 5;
  const int wm = w >> 1, wn = w & 1;   // 4 x 2 wave grid
  const int n0 = blockIdx.x * 128;
  const int m0 = blockIdx.y * 128;
  const int nk = K >> 5;

  auto stage = [&](int kt, int buf) {
    const int kbase = kt * 32;
#pragma unroll
    for (int i = 0; i < 2; ++i) {
      int c = t + i * 256;         // 512 chunks of 16B per 128x32 tile
      int row = c >> 2, seg = c & 3;
      const bf16* ga = A + (size_t)(m0 + row) * K + kbase + seg * 8;
      const bf16* gb = Wt + (size_t)(n0 + row) * K + kbase + seg * 8;
      ASYNC_B128(ga, &sA[buf][row * LD + seg * 8]);
      ASYNC_B128(gb, &sB[buf][row * LD + seg * 8]);
    }
  };

  v8f acc[2][4];
#pragma unroll
  for (int i = 0; i < 2; ++i)
#pragma unroll
    for (int j = 0; j < 4; ++j) acc[i][j] = v8f_zero();

  stage(0, 0);
  for (int kt = 0; kt < nk; ++kt) {
    const int cur = kt & 1;
    if (kt + 1 < nk) {
      stage(kt + 1, cur ^ 1);
      ASYNC_WAIT(4);
    } else {
      ASYNC_WAIT(0);
    }
    __syncthreads();

    v16bf afr[2], bfr[4];
#pragma unroll
    for (int fm = 0; fm < 2; ++fm)
      afr[fm] = load_frag_a(sA[cur], wm * 32 + fm * 16, 0, LD);
#pragma unroll
    for (int fn = 0; fn < 4; ++fn)
      bfr[fn] = load_frag_b(sB[cur], wn * 64 + fn * 16, 0, LD);
#pragma unroll
    for (int fm = 0; fm < 2; ++fm)
#pragma unroll
      for (int fn = 0; fn < 4; ++fn)
        acc[fm][fn] = wmma_bf16(afr[fm], bfr[fn], acc[fm][fn]);
    __syncthreads();
  }

  const int l = lane & 15, hf = lane >> 4;
#pragma unroll
  for (int fm = 0; fm < 2; ++fm)
#pragma unroll
    for (int fn = 0; fn < 4; ++fn)
#pragma unroll
      for (int r = 0; r < 8; ++r) {
        int row = m0 + wm * 32 + fm * 16 + r + hf * 8;
        int col = n0 + wn * 64 + fn * 16 + l;
        float v = acc[fm][fn][r] + bias[col];
        if (act) v = 0.5f * v * (1.0f + erff(v * 0.70710678118654752f));
        size_t idx = (size_t)row * N + col;
        if (outF) outF[idx] = v;
        if (outB) outB[idx] = (bf16)v;
      }
}

// ---------------- flash-style attention ----------------
// grid (SEQ/64, NHEAD, BATCH), 128 threads (4 waves). One wave = 16 query rows.
__global__ __launch_bounds__(128) void attn_wmma(const bf16* __restrict__ Q,
                                                 const bf16* __restrict__ Kb,
                                                 const bf16* __restrict__ V,
                                                 bf16* __restrict__ ctx) {
  constexpr int LD = 72;  // 64 + 8 pad
  __shared__ __attribute__((aligned(16))) bf16 sQ[64 * LD];
  __shared__ __attribute__((aligned(16))) bf16 sK[64 * LD];
  __shared__ __attribute__((aligned(16))) bf16 sVt[64 * LD];  // [dh][key]
  __shared__ __attribute__((aligned(16))) bf16 sP[64 * LD];

  const int t = (int)threadIdx.x, lane = t & 31, w = t >> 5;
  const int l = lane & 15, hf = lane >> 4;
  const int b = blockIdx.z, h = blockIdx.y;
  const int q0 = blockIdx.x * 64;

  const bf16* Qbase = Q + (size_t)b * SEQ * DMODEL + (size_t)h * DHEAD;
  const bf16* Kbase = Kb + (size_t)b * SEQ * DMODEL + (size_t)h * DHEAD;
  const bf16* Vbase = V + (size_t)b * SEQ * DMODEL + (size_t)h * DHEAD;

  // stage Q tile (64 rows x 64 dh = 128B/row -> 512 b128 chunks)
#pragma unroll
  for (int i = 0; i < 4; ++i) {
    int c = t + i * 128;
    int row = c >> 3, seg = c & 7;
    ASYNC_B128(Qbase + (size_t)(q0 + row) * DMODEL + seg * 8,
               &sQ[row * LD + seg * 8]);
  }
  ASYNC_WAIT(0);
  __syncthreads();

  v16bf aq0 = load_frag_a(sQ, w * 16, 0, LD);
  v16bf aq1 = load_frag_a(sQ, w * 16, 32, LD);

  float mrow[8], lrow[8];
  v8f o[4];
#pragma unroll
  for (int r = 0; r < 8; ++r) { mrow[r] = -1e30f; lrow[r] = 0.0f; }
#pragma unroll
  for (int dn = 0; dn < 4; ++dn) o[dn] = v8f_zero();

  for (int kc = 0; kc < SEQ / 64; ++kc) {
    const int k0 = kc * 64;
    // stage K chunk (async)
#pragma unroll
    for (int i = 0; i < 4; ++i) {
      int c = t + i * 128;
      int row = c >> 3, seg = c & 7;
      ASYNC_B128(Kbase + (size_t)(k0 + row) * DMODEL + seg * 8,
                 &sK[row * LD + seg * 8]);
    }
    // stage V transposed: sVt[dh][key]
    {
      int key = t & 63, half = t >> 6;  // half selects dh 0..31 / 32..63
      union { u32x4 q[4]; bf16 e[32]; } u;
      const bf16* g = Vbase + (size_t)(k0 + key) * DMODEL + half * 32;
      u.q[0] = ((const u32x4*)g)[0];
      u.q[1] = ((const u32x4*)g)[1];
      u.q[2] = ((const u32x4*)g)[2];
      u.q[3] = ((const u32x4*)g)[3];
#pragma unroll
      for (int j = 0; j < 32; ++j) sVt[(half * 32 + j) * LD + key] = u.e[j];
    }
    ASYNC_WAIT(0);
    __syncthreads();

    // S = Q K^T * (1/sqrt(DH))
    v8f sfr[4];
#pragma unroll
    for (int n = 0; n < 4; ++n) {
      v8f a = v8f_zero();
      v16bf bk0 = load_frag_b(sK, n * 16, 0, LD);
      v16bf bk1 = load_frag_b(sK, n * 16, 32, LD);
      a = wmma_bf16(aq0, bk0, a);
      a = wmma_bf16(aq1, bk1, a);
      sfr[n] = a * 0.125f;
    }

    // online softmax; row r lives in VGPR r, half-wave lanes = keys
#pragma unroll
    for (int r = 0; r < 8; ++r) {
      float mx = fmaxf(fmaxf(sfr[0][r], sfr[1][r]), fmaxf(sfr[2][r], sfr[3][r]));
#pragma unroll
      for (int off = 1; off < 16; off <<= 1)
        mx = fmaxf(mx, __shfl_xor(mx, off, 32));
      float mnew = fmaxf(mrow[r], mx);
      float corr = __expf(mrow[r] - mnew);
      mrow[r] = mnew;
      float rs = 0.0f;
#pragma unroll
      for (int n = 0; n < 4; ++n) {
        float p = __expf(sfr[n][r] - mnew);
        rs += p;
        sP[(w * 16 + r + hf * 8) * LD + n * 16 + l] = (bf16)p;
      }
#pragma unroll
      for (int off = 1; off < 16; off <<= 1) rs += __shfl_xor(rs, off, 32);
      lrow[r] = lrow[r] * corr + rs;
#pragma unroll
      for (int dn = 0; dn < 4; ++dn) o[dn][r] = o[dn][r] * corr;
    }
    __syncthreads();

    // O += P V
    v16bf pa0 = load_frag_a(sP, w * 16, 0, LD);
    v16bf pa1 = load_frag_a(sP, w * 16, 32, LD);
#pragma unroll
    for (int dn = 0; dn < 4; ++dn) {
      v16bf bv0 = load_frag_b(sVt, dn * 16, 0, LD);
      v16bf bv1 = load_frag_b(sVt, dn * 16, 32, LD);
      o[dn] = wmma_bf16(pa0, bv0, o[dn]);
      o[dn] = wmma_bf16(pa1, bv1, o[dn]);
    }
    __syncthreads();
  }

  bf16* cb = ctx + (size_t)b * SEQ * DMODEL + (size_t)h * DHEAD;
#pragma unroll
  for (int dn = 0; dn < 4; ++dn)
#pragma unroll
    for (int r = 0; r < 8; ++r) {
      int qr = q0 + w * 16 + r + hf * 8;
      int dh = dn * 16 + l;
      cb[(size_t)qr * DMODEL + dh] = (bf16)(o[dn][r] / lrow[r]);
    }
}

// ---------------- LayerNorm + residual ----------------
__global__ __launch_bounds__(256) void ln_residual(
    const float* __restrict__ xin, const float* __restrict__ res,
    const float* __restrict__ gw, const float* __restrict__ bw,
    float* __restrict__ outF, bf16* __restrict__ outB) {
  const int row = blockIdx.x, t = (int)threadIdx.x, lane = t & 31, w = t >> 5;
  const size_t base = (size_t)row * DMODEL;
  __shared__ float red[8];

  f32x4 xv = *(const f32x4*)(xin + base + t * 4);
  float s = xv[0] + xv[1] + xv[2] + xv[3];
#pragma unroll
  for (int off = 1; off < 32; off <<= 1) s += __shfl_xor(s, off, 32);
  if (lane == 0) red[w] = s;
  __syncthreads();
  float tot = 0.0f;
#pragma unroll
  for (int i = 0; i < 8; ++i) tot += red[i];
  const float mu = tot * (1.0f / DMODEL);

  float d0 = xv[0] - mu, d1 = xv[1] - mu, d2 = xv[2] - mu, d3 = xv[3] - mu;
  float s2 = d0 * d0 + d1 * d1 + d2 * d2 + d3 * d3;
#pragma unroll
  for (int off = 1; off < 32; off <<= 1) s2 += __shfl_xor(s2, off, 32);
  __syncthreads();
  if (lane == 0) red[w] = s2;
  __syncthreads();
  float tot2 = 0.0f;
#pragma unroll
  for (int i = 0; i < 8; ++i) tot2 += red[i];
  const float rstd = rsqrtf(tot2 * (1.0f / DMODEL) + 1e-5f);

  f32x4 rv = *(const f32x4*)(res + base + t * 4);
  f32x4 gv = *(const f32x4*)(gw + t * 4);
  f32x4 bv = *(const f32x4*)(bw + t * 4);
  f32x4 ov;
  ov[0] = d0 * rstd * gv[0] + bv[0] + rv[0];
  ov[1] = d1 * rstd * gv[1] + bv[1] + rv[1];
  ov[2] = d2 * rstd * gv[2] + bv[2] + rv[2];
  ov[3] = d3 * rstd * gv[3] + bv[3] + rv[3];
  *(f32x4*)(outF + base + t * 4) = ov;
  if (outB) {
    union { bf16 e[4]; unsigned long long u; } ob;
#pragma unroll
    for (int j = 0; j < 4; ++j) ob.e[j] = (bf16)ov[j];
    *(unsigned long long*)(outB + base + t * 4) = ob.u;
  }
}

// ---------------- host launcher ----------------
extern "C" void kernel_launch(void* const* d_in, const int* in_sizes, int n_in,
                              void* d_out, int out_size, void* d_ws, size_t ws_size,
                              hipStream_t stream) {
  const float* x   = (const float*)d_in[0];
  const float* Wq  = (const float*)d_in[1];
  const float* bq  = (const float*)d_in[2];
  const float* Wk  = (const float*)d_in[3];
  const float* bk  = (const float*)d_in[4];
  const float* Wv  = (const float*)d_in[5];
  const float* bv  = (const float*)d_in[6];
  const float* Wo  = (const float*)d_in[7];
  const float* bo  = (const float*)d_in[8];
  const float* g1  = (const float*)d_in[9];
  const float* b1  = (const float*)d_in[10];
  const float* Wf1 = (const float*)d_in[11];
  const float* bf1 = (const float*)d_in[12];
  const float* Wf2 = (const float*)d_in[13];
  const float* bf2 = (const float*)d_in[14];
  const float* g2  = (const float*)d_in[15];
  const float* b2  = (const float*)d_in[16];

  char* p = (char*)d_ws;
  auto take = [&](size_t bytes) -> char* {
    char* r = p;
    p += (bytes + 255) & ~(size_t)255;
    return r;
  };
  const size_t MD2 = (size_t)MROWS * DMODEL * sizeof(bf16);
  const size_t MD4 = (size_t)MROWS * DMODEL * sizeof(float);

  bf16* xb   = (bf16*)take(MD2);
  bf16* wqt  = (bf16*)take((size_t)DMODEL * DMODEL * 2);
  bf16* wkt  = (bf16*)take((size_t)DMODEL * DMODEL * 2);
  bf16* wvt  = (bf16*)take((size_t)DMODEL * DMODEL * 2);
  bf16* wot  = (bf16*)take((size_t)DMODEL * DMODEL * 2);
  bf16* wf1t = (bf16*)take((size_t)DMODEL * FFDIM * 2);
  bf16* wf2t = (bf16*)take((size_t)FFDIM * DMODEL * 2);
  bf16* qb   = (bf16*)take(MD2);
  bf16* kb   = (bf16*)take(MD2);
  bf16* vb   = (bf16*)take(MD2);
  bf16* ctx  = (bf16*)take(MD2);
  float* x1  = (float*)take(MD4);
  float* x2  = (float*)take(MD4);
  bf16* hbuf = (bf16*)take((size_t)MROWS * FFDIM * 2);
  bf16* x2b  = xb;  // reuse: xb dead after QKV projections
  float* x3  = x1;  // reuse: x1 dead after first LN

  // 1) conversions
  {
    int n = MROWS * DMODEL;
    cvt_bf16_kernel<<<n / 4 / 256, 256, 0, stream>>>(x, xb, n);
    cvt_t_kernel<<<(DMODEL * DMODEL) / 256, 256, 0, stream>>>(Wq, wqt, DMODEL, DMODEL);
    cvt_t_kernel<<<(DMODEL * DMODEL) / 256, 256, 0, stream>>>(Wk, wkt, DMODEL, DMODEL);
    cvt_t_kernel<<<(DMODEL * DMODEL) / 256, 256, 0, stream>>>(Wv, wvt, DMODEL, DMODEL);
    cvt_t_kernel<<<(DMODEL * DMODEL) / 256, 256, 0, stream>>>(Wo, wot, DMODEL, DMODEL);
    cvt_t_kernel<<<(DMODEL * FFDIM) / 256, 256, 0, stream>>>(Wf1, wf1t, DMODEL, FFDIM);
    cvt_t_kernel<<<(FFDIM * DMODEL) / 256, 256, 0, stream>>>(Wf2, wf2t, FFDIM, DMODEL);
  }

  // 2) QKV projections (bf16 out, bias fused)
  dim3 gDD(DMODEL / 128, MROWS / 128);
  gemm_bf16_wmma<<<gDD, 256, 0, stream>>>(xb, wqt, bq, nullptr, qb, MROWS, DMODEL, DMODEL, 0);
  gemm_bf16_wmma<<<gDD, 256, 0, stream>>>(xb, wkt, bk, nullptr, kb, MROWS, DMODEL, DMODEL, 0);
  gemm_bf16_wmma<<<gDD, 256, 0, stream>>>(xb, wvt, bv, nullptr, vb, MROWS, DMODEL, DMODEL, 0);

  // 3) attention (flash-style, no LxL materialization)
  attn_wmma<<<dim3(SEQ / 64, NHEAD, BATCH), 128, 0, stream>>>(qb, kb, vb, ctx);

  // 4) output projection -> x1 (fp32)
  gemm_bf16_wmma<<<gDD, 256, 0, stream>>>(ctx, wot, bo, x1, nullptr, MROWS, DMODEL, DMODEL, 0);

  // 5) x2 = LN(x1)*g1+b1 + x  (fp32 + bf16 copy for FFN)
  ln_residual<<<MROWS, 256, 0, stream>>>(x1, x, g1, b1, x2, x2b);

  // 6) FFN1 with fused exact GELU -> h (bf16)
  gemm_bf16_wmma<<<dim3(FFDIM / 128, MROWS / 128), 256, 0, stream>>>(
      x2b, wf1t, bf1, nullptr, hbuf, MROWS, FFDIM, DMODEL, 1);

  // 7) FFN2 -> x3 (fp32)
  gemm_bf16_wmma<<<gDD, 256, 0, stream>>>(hbuf, wf2t, bf2, x3, nullptr, MROWS, DMODEL, FFDIM, 0);

  // 8) out = LN(x3)*g2+b2 + x2
  ln_residual<<<MROWS, 256, 0, stream>>>(x3, x2, g2, b2, (float*)d_out, nullptr);

  (void)in_sizes; (void)n_in; (void)out_size; (void)ws_size;
}
